// AdvancedGraphNeuralNetwork_5179730559068
// MI455X (gfx1250) — matrix operand order repaired and checked by reading, not compiled, not run on hardware
//
#include <hip/hip_runtime.h>
#include <hip/hip_bf16.h>

typedef __attribute__((ext_vector_type(16))) _Float16 v16h;
typedef __attribute__((ext_vector_type(8)))  _Float16 v8h;
typedef __attribute__((ext_vector_type(8)))  float    v8f;

#define NB    32
#define SS    512
#define NN    24
#define HH    64
#define SH_ST 65   // f32 h master copy stride (dwords)
#define T_ST  40   // f16 stride (halves) for h16 / att16 / WhT  (80B = 5*16B; 20-dword stride -> conflict-free)
#define W_ST  72   // f16 stride (halves) for transposed W       (144B = 9*16B; 36-dword stride -> conflict-free)

// Fragment built from two contiguous 8-half (16B) chunks, `gap` halves apart.
// A-frag (row-major src):   base = row + k0 + hi*8,  gap = 16
// B-frag (transposed src):  base = col*stride + k0 + hi*8,  gap = 16
__device__ __forceinline__ v16h frag2x8(const _Float16* p, int gap) {
  v8h lo = *(const v8h*)p;
  v8h hi = *(const v8h*)(p + gap);
  return __builtin_shufflevector(lo, hi, 0, 1, 2, 3, 4, 5, 6, 7,
                                 8, 9, 10, 11, 12, 13, 14, 15);
}

// ---------------- Kernel A: fused 3-layer GAT + temporal accumulation ----------------
// grid = NB * NC blocks, 128 threads (4 waves). Block (b, chunk) handles SC=S/NC graphs.
__global__ __launch_bounds__(128) void gat_fused_kernel(
    const float* __restrict__ x, const float* __restrict__ node_emb,
    const float* __restrict__ W, const float* __restrict__ avec,
    float* __restrict__ part, float* __restrict__ corr0, float* __restrict__ corr1,
    int NC, int SC) {
  __shared__ float sh[32 * SH_ST];                      // h, f32 master (residual + pooling)
  __shared__ __align__(16) _Float16 sH16[32 * T_ST];    // h, f16 shadow (A-frags)
  __shared__ __align__(16) _Float16 sWhT[64 * T_ST];    // Wh transposed: [feat][node], f16 (B-frags, f1/f2)
  __shared__ __align__(16) _Float16 sAtt[32 * T_ST];    // att, f16 row-major (A-frags)
  __shared__ __align__(16) _Float16 sWT[3 * 64 * W_ST]; // W transposed per layer: [out][in], f16
  __shared__ float sA1[3 * 64], sA2[3 * 64];
  __shared__ float sF1[32], sF2[32];

  const int tid  = threadIdx.x;
  const int wave = tid >> 5;
  const int lane = tid & 31;
  const int ln   = lane & 15;
  const int hi   = lane >> 4;
  const int b     = blockIdx.x / NC;
  const int chunk = blockIdx.x % NC;

  // hoist W (transposed, f16) and attention vectors into LDS once
  for (int e = tid; e < 3 * 64 * 64; e += 128) {
    int layer = e >> 12, r = (e >> 6) & 63, c = e & 63;  // W[layer][r=in][c=out]
    sWT[(layer * 64 + c) * W_ST + r] = (_Float16)W[e];
  }
  for (int e = tid; e < 3 * 128; e += 128) {
    int layer = e >> 7, j = e & 127;
    if (j < 64) sA1[layer * 64 + j] = avec[e];
    else        sA2[layer * 64 + (j - 64)] = avec[e];
  }
  float acc[12];
#pragma unroll
  for (int q = 0; q < 12; ++q) acc[q] = 0.f;
  __syncthreads();

  for (int si = 0; si < SC; ++si) {
    const int s = chunk * SC + si;
    const int g = b * SS + s;

    // build h[n][k] = x[g,n] * node_emb[n,k]; rows 24..31 zero (f32 + f16 shadow)
    for (int e = tid; e < 32 * 64; e += 128) {
      int n = e >> 6, k = e & 63;
      float v = (n < NN) ? x[g * NN + n] * node_emb[n * HH + k] : 0.f;
      sh[n * SH_ST + k]  = v;
      sH16[n * T_ST + k] = (_Float16)v;
    }
    __syncthreads();

    for (int layer = 0; layer < 3; ++layer) {
      const _Float16* Wl = &sWT[layer * 64 * W_ST];

      // ---- Wh = h(32x64) @ W(64x64): 8 tiles, 2 per wave, K=64 (2 wmma steps)
#pragma unroll
      for (int j2 = 0; j2 < 2; ++j2) {
        const int t  = wave * 2 + j2;
        const int m0 = (t >> 2) * 16, n0 = (t & 3) * 16;
        v8f c = {};
#pragma unroll
        for (int kk = 0; kk < 2; ++kk) {
          v16h af = frag2x8(sH16 + (m0 + ln) * T_ST + kk * 32 + hi * 8, 16);
          v16h bf = frag2x8(Wl + (n0 + ln) * W_ST + kk * 32 + hi * 8, 16);
          c = __builtin_amdgcn_wmma_f32_16x16x32_f16(false, af, false, bf,
                                                     (short)0, c, false, false);
        }
        // store transposed f16: WhT[feat = n0+ln][node = m0+hi*8 + r], 8 contiguous halves
        v8h o;
#pragma unroll
        for (int r = 0; r < 8; ++r) o[r] = (_Float16)c[r];
        *(v8h*)(sWhT + (n0 + ln) * T_ST + m0 + hi * 8) = o;
      }
      __syncthreads();

      // ---- f1, f2 and zero att (f16)
      for (int e = tid; e < (32 * T_ST) / 2; e += 128) ((unsigned*)sAtt)[e] = 0u;
      if (tid < NN) {
        float s1 = 0.f;
        for (int k = 0; k < HH; ++k)
          s1 += (float)sWhT[k * T_ST + tid] * sA1[layer * 64 + k];
        sF1[tid] = s1;
      } else if (tid >= 64 && tid < 64 + NN) {
        int n = tid - 64;
        float s2 = 0.f;
        for (int k = 0; k < HH; ++k)
          s2 += (float)sWhT[k * T_ST + n] * sA2[layer * 64 + k];
        sF2[n] = s2;
      }
      __syncthreads();

      // ---- row softmax of leaky_relu(f1[n]+f2[j])  (sigmoid mask always true)
      if (tid < NN) {
        const float f1 = sF1[tid];
        float mx = -1e30f;
        float ev[NN];
#pragma unroll
        for (int j = 0; j < NN; ++j) {
          float e0 = f1 + sF2[j];
          e0 = (e0 > 0.f) ? e0 : 0.2f * e0;
          ev[j] = e0;
          mx = fmaxf(mx, e0);
        }
        float sum = 0.f;
#pragma unroll
        for (int j = 0; j < NN; ++j) { ev[j] = __expf(ev[j] - mx); sum += ev[j]; }
        const float inv = 1.f / sum;
#pragma unroll
        for (int j = 0; j < NN; ++j) sAtt[tid * T_ST + j] = (_Float16)(ev[j] * inv);
      }
      __syncthreads();

      // ---- h += elu( att(32x32) @ Wh(32x64) ): 8 tiles, K=32 (1 wmma step)
      v8f hf[2];
#pragma unroll
      for (int j2 = 0; j2 < 2; ++j2) {
        const int t  = wave * 2 + j2;
        const int m0 = (t >> 2) * 16, n0 = (t & 3) * 16;
        v16h af = frag2x8(sAtt + (m0 + ln) * T_ST + hi * 8, 16);
        v16h bf = frag2x8(sWhT + (n0 + ln) * T_ST + hi * 8, 16);
        v8f c = {};
        c = __builtin_amdgcn_wmma_f32_16x16x32_f16(false, af, false, bf,
                                                   (short)0, c, false, false);
        hf[j2] = c;
      }
#pragma unroll
      for (int j2 = 0; j2 < 2; ++j2) {
        const int t  = wave * 2 + j2;
        const int m0 = (t >> 2) * 16, n0 = (t & 3) * 16;
        const int n = n0 + ln;
#pragma unroll
        for (int r = 0; r < 8; ++r) {
          float v  = hf[j2][r];
          float el = (v > 0.f) ? v : (__expf(v) - 1.f);  // elu; rows>=24 get +0
          const int m = m0 + hi * 8 + r;
          float nv = sh[m * SH_ST + n] + el;
          sh[m * SH_ST + n]  = nv;
          sH16[m * T_ST + n] = (_Float16)nv;
        }
      }
      __syncthreads();
    }

    // accumulate pooling sums (register-resident, deterministic)
#pragma unroll
    for (int q = 0; q < 12; ++q) {
      int e = tid * 12 + q;
      acc[q] += sh[(e >> 6) * SH_ST + (e & 63)];
    }
    // boundary corrections for the collapsed conv+mean
    if (s == 0      && tid < HH) corr0[b * HH + tid] = sh[0 * SH_ST + tid];
    if (s == SS - 1 && tid < HH) corr1[b * HH + tid] = sh[23 * SH_ST + tid];
    __syncthreads();
  }

#pragma unroll
  for (int q = 0; q < 12; ++q)
    part[(size_t)blockIdx.x * (NN * HH) + tid * 12 + q] = acc[q];
}

// ---------------- Kernel B: reduce + collapsed conv/pool + MLP head ----------------
__global__ __launch_bounds__(128) void tail_kernel(
    const float* __restrict__ part, const float* __restrict__ corr0,
    const float* __restrict__ corr1, const float* __restrict__ conv_w,
    const float* __restrict__ conv_b, const float* __restrict__ out1_w,
    const float* __restrict__ out1_b, const float* __restrict__ ln_g,
    const float* __restrict__ ln_b, const float* __restrict__ out2_w,
    const float* __restrict__ out2_b, float* __restrict__ out, int NC) {
  __shared__ float P[NN * HH];      // P[n*64+i] = mean_s h_final[b,s,n,i]
  __shared__ float PL0[HH], PR23[HH];
  __shared__ float flat[HH * NN];   // flat[o*24+n]
  __shared__ float z[HH], stats[2];
  const int b = blockIdx.x, tid = threadIdx.x;

  for (int e = tid; e < NN * HH; e += 128) {
    float s = 0.f;
    for (int c = 0; c < NC; ++c) s += part[((size_t)(b * NC + c)) * (NN * HH) + e];
    P[e] = s * (1.f / (float)SS);
  }
  __syncthreads();
  if (tid < HH) {
    PL0[tid]  = P[23 * HH + tid] - corr1[b * HH + tid] * (1.f / (float)SS);
    PR23[tid] = P[0 * HH + tid]  - corr0[b * HH + tid] * (1.f / (float)SS);
  }
  __syncthreads();

  // pooled[o][n] = conv_b[o] + sum_i cw[o,i,0]*L + cw[o,i,1]*P + cw[o,i,2]*R
  for (int e = tid; e < HH * NN; e += 128) {
    const int o = e / NN, n = e % NN;
    float s = conv_b[o];
    for (int i = 0; i < HH; ++i) {
      float left  = (n >= 1)      ? P[(n - 1) * HH + i] : PL0[i];
      float mid   = P[n * HH + i];
      float right = (n <= NN - 2) ? P[(n + 1) * HH + i] : PR23[i];
      const float* cw = conv_w + (o * HH + i) * 3;
      s += cw[0] * left + cw[1] * mid + cw[2] * right;
    }
    flat[o * NN + n] = s;
  }
  __syncthreads();

  if (tid < HH) {
    float s = out1_b[tid];
    for (int e = 0; e < HH * NN; ++e) s += flat[e] * out1_w[e * HH + tid];
    z[tid] = s;
  }
  __syncthreads();
  if (tid == 0) {
    float mu = 0.f;
    for (int j = 0; j < HH; ++j) mu += z[j];
    mu *= (1.f / HH);
    float var = 0.f;
    for (int j = 0; j < HH; ++j) { float d = z[j] - mu; var += d * d; }
    stats[0] = mu; stats[1] = var * (1.f / HH);
  }
  __syncthreads();
  if (tid < HH) {
    float zn = (z[tid] - stats[0]) * rsqrtf(stats[1] + 1e-5f) * ln_g[tid] + ln_b[tid];
    z[tid] = zn > 0.f ? zn : 0.f;
  }
  __syncthreads();
  if (tid == 0) {
    float s = out2_b[0];
    for (int j = 0; j < HH; ++j) s += z[j] * out2_w[j];
    out[b] = s;
  }
}

extern "C" void kernel_launch(void* const* d_in, const int* in_sizes, int n_in,
                              void* d_out, int out_size, void* d_ws, size_t ws_size,
                              hipStream_t stream) {
  const float* x        = (const float*)d_in[0];
  // d_in[1] = adj_matrix: dead (sigmoid > 0 always true)
  const float* node_emb = (const float*)d_in[2];
  const float* W        = (const float*)d_in[3];
  const float* avec     = (const float*)d_in[4];
  const float* conv_w   = (const float*)d_in[5];
  const float* conv_b   = (const float*)d_in[6];
  const float* out1_w   = (const float*)d_in[7];
  const float* out1_b   = (const float*)d_in[8];
  const float* ln_g     = (const float*)d_in[9];
  const float* ln_b     = (const float*)d_in[10];
  const float* out2_w   = (const float*)d_in[11];
  const float* out2_b   = (const float*)d_in[12];
  float* out = (float*)d_out;

  // pick chunks-per-batch from available scratch: need NB*NC*1536 + 2*NB*64 floats
  int NC = 64;
  while (NC > 1) {
    size_t need = ((size_t)NB * NC * (NN * HH) + 2u * NB * HH) * sizeof(float);
    if (need <= ws_size) break;
    NC >>= 2;
  }
  const int SC = SS / NC;

  float* part  = (float*)d_ws;
  float* corr0 = part + (size_t)NB * NC * (NN * HH);
  float* corr1 = corr0 + NB * HH;

  gat_fused_kernel<<<NB * NC, 128, 0, stream>>>(x, node_emb, W, avec,
                                                part, corr0, corr1, NC, SC);
  tail_kernel<<<NB, 128, 0, stream>>>(part, corr0, corr1, conv_w, conv_b,
                                      out1_w, out1_b, ln_g, ln_b,
                                      out2_w, out2_b, out, NC);
}
